// MeshGNN_88682484727942
// MI455X (gfx1250) — matrix operand order repaired and checked by reading, not compiled, not run on hardware
//
#include <hip/hip_runtime.h>

// ---------------------------------------------------------------------------
// MeshGNN for MI455X (gfx1250): fully-fused bf16 WMMA implementation.
//   grid: 2048 blocks x 384 threads (12 wave32), 16 batch rows per block.
//   h (16 x 12 x 256, bf16) lives entirely in LDS for all 4 GNN layers.
// ---------------------------------------------------------------------------

typedef __bf16 bf16_t;
typedef __attribute__((ext_vector_type(16))) __bf16 v16bf;
typedef __attribute__((ext_vector_type(8)))  __bf16 v8bf;
typedef __attribute__((ext_vector_type(8)))  float  v8f;

#define BATCH      32768
#define TEXT_DIM   384
#define HIDDEN     256
#define NVERTS     12
#define MT         16                 // batch rows per workgroup
#define NWAVES     12                 // one wave per WMMA M-tile (MT*12/16)
#define BLOCK_THR  (NWAVES * 32)

// ws layout (bytes)
#define WS_TEXTBF_OFF  0u
#define WS_TEXTBF_SZ   (BATCH * TEXT_DIM * 2u)                 // 25,165,824
#define WS_WTEXT_OFF   (WS_TEXTBF_OFF + WS_TEXTBF_SZ)
#define WS_WTEXT_SZ    (TEXT_DIM * HIDDEN * 2u)                //    196,608
#define WS_WGNN_OFF    (WS_WTEXT_OFF + WS_WTEXT_SZ)
#define WS_WGNN_SZ     (4u * HIDDEN * HIDDEN * 2u)             //    524,288

// ---------------------------------------------------------------------------
// Kernel 1: fp32 -> bf16 conversion of text embeddings (vectorized x4).
// ---------------------------------------------------------------------------
__global__ void cvt_text_kernel(const float* __restrict__ in,
                                bf16_t* __restrict__ out, int n4) {
  int i = blockIdx.x * blockDim.x + threadIdx.x;
  if (i < n4) {
    float4 v = ((const float4*)in)[i];
    union { v8bf dummy; struct { bf16_t a, b, c, d; } s; } u;   // 16B store
    u.s.a = (bf16_t)v.x; u.s.b = (bf16_t)v.y;
    u.s.c = (bf16_t)v.z; u.s.d = (bf16_t)v.w;
    ((decltype(u)*)out)[i] = u;
  }
}

// ---------------------------------------------------------------------------
// Kernel 2: swizzle W_text / W_gnn into WMMA B-fragment layout.
// B 32x16 bf16 tile (ISA 7.12.2): lane L holds column n = L%16,
// element j <-> K = (L<16 ? 0 : 8) + (j&7) + (j<8 ? 0 : 16). Stored per-lane contiguous (32B).
// flat element index e = ((tk*16 + tn)*32 + lane)*16 + j.
// ---------------------------------------------------------------------------
__global__ void swizzle_w_kernel(const float* __restrict__ Wt,
                                 const float* __restrict__ Wg,
                                 bf16_t* __restrict__ wt_sw,
                                 bf16_t* __restrict__ wg_sw) {
  int tid = blockIdx.x * blockDim.x + threadIdx.x;
  const int NTEXT = TEXT_DIM * HIDDEN;          // 98304 elements
  const int NGNN  = 4 * HIDDEN * HIDDEN;        // 262144 elements
  if (tid < NTEXT) {
    int e = tid;
    int j = e & 15, lane = (e >> 4) & 31, tile = e >> 9;
    int tn = tile & 15, tk = tile >> 4;         // 12 K-tiles x 16 N-tiles
    int k = tk * 32 + ((lane & 16) ? 8 : 0) + (j & 7) + ((j & 8) ? 16 : 0);
    int n = tn * 16 + (lane & 15);
    wt_sw[e] = (bf16_t)Wt[k * HIDDEN + n];
  } else if (tid < NTEXT + NGNN) {
    int e = tid - NTEXT;
    int l = e >> 16, e2 = e & 65535;
    int j = e2 & 15, lane = (e2 >> 4) & 31, tile = e2 >> 9;
    int tn = tile & 15, tk = tile >> 4;         // 8 K-tiles x 16 N-tiles
    int k = tk * 32 + ((lane & 16) ? 8 : 0) + (j & 7) + ((j & 8) ? 16 : 0);
    int n = tn * 16 + (lane & 15);
    wg_sw[e] = (bf16_t)Wg[l * 65536 + k * HIDDEN + n];
  }
}

// ---------------------------------------------------------------------------
// Kernel 3: fused text-proj + 4 GNN layers + output head.
// ---------------------------------------------------------------------------
__launch_bounds__(BLOCK_THR, 1)
__global__ void mesh_gnn_kernel(const bf16_t* __restrict__ textbf,
                                const bf16_t* __restrict__ wt_sw,
                                const bf16_t* __restrict__ wg_sw,
                                const float* __restrict__ b_text,
                                const float* __restrict__ b_gnn,
                                const float* __restrict__ W_out,
                                const float* __restrict__ b_out,
                                const float* __restrict__ adjacency,
                                const float* __restrict__ templ,
                                float* __restrict__ out) {
  __shared__ bf16_t h_lds[MT * NVERTS * HIDDEN];   // 96 KB, row r = b*12+n
  __shared__ bf16_t tf_lds[MT * HIDDEN];           //  8 KB
  __shared__ float  s_rowsum[NVERTS];

  const int tid  = threadIdx.x;
  const int wave = tid >> 5;
  const int lane = tid & 31;
  const int l15  = lane & 15;
  const int khA  = (lane & 16) ? 8 : 0;            // A-frag K-half (16-bit A layout)
  const int rowh = (lane & 16) ? 8 : 0;            // C-frag M-half
  const int bbase = blockIdx.x * MT;

  // adjacency row for the (b,n) row this lane owns in the GNN phases
  const int r_own = wave * 16 + l15;
  const int n_own = r_own % NVERTS;
  const int b_own = r_own / NVERTS;
  float adjv[NVERTS];
#pragma unroll
  for (int m = 0; m < NVERTS; ++m) adjv[m] = adjacency[n_own * NVERTS + m];

  if (tid < NVERTS) {
    float s = 0.f;
#pragma unroll
    for (int m = 0; m < NVERTS; ++m) s += adjacency[tid * NVERTS + m];
    s_rowsum[tid] = s;
  }

  // keep GNN weights flowing into L2 early
  __builtin_prefetch(wg_sw, 0, 1);

  // ---- Phase 0: tf = text @ W_text + b_text (M = 16 batch rows) ----------
  if (wave < 8) {
#pragma unroll
    for (int t = 0; t < 2; ++t) {
      const int nt = wave * 2 + t;
      v8f acc = {};
#pragma unroll
      for (int kt = 0; kt < TEXT_DIM / 32; ++kt) {
        const bf16_t* ap = textbf + (size_t)(bbase + l15) * TEXT_DIM + kt * 32 + khA;
        union { v16bf v; v8bf h[2]; } ua;
        ua.h[0] = *(const v8bf*)ap;          // K = kh + 0..7
        ua.h[1] = *(const v8bf*)(ap + 16);   // K = kh + 16..23
        const v16bf bf = *(const v16bf*)(wt_sw + ((size_t)(kt * 16 + nt) * 32 + lane) * 16);
        acc = __builtin_amdgcn_wmma_f32_16x16x32_bf16(false, ua.v, false, bf,
                                                      (short)0, acc, false, false);
      }
      const int col = nt * 16 + l15;
      const float bias = b_text[col];
#pragma unroll
      for (int v = 0; v < 8; ++v)
        tf_lds[(v + rowh) * HIDDEN + col] = (bf16_t)(acc[v] + bias);
    }
  }
  __syncthreads();

  // ---- Phase 1: layer 0 collapses (h is vertex-broadcast):
  //      h1[b,n] = relu(rowsum[n] * (tf @ W0)[b] + b0) ----------------------
  if (wave < 8) {
#pragma unroll
    for (int t = 0; t < 2; ++t) {
      const int nt = wave * 2 + t;
      v8f acc = {};
#pragma unroll
      for (int kt = 0; kt < HIDDEN / 32; ++kt) {
        const bf16_t* ap = tf_lds + l15 * HIDDEN + kt * 32 + khA;
        union { v16bf v; v8bf h[2]; } ua;
        ua.h[0] = *(const v8bf*)ap;
        ua.h[1] = *(const v8bf*)(ap + 16);
        const v16bf bf = *(const v16bf*)(wg_sw + ((size_t)(kt * 16 + nt) * 32 + lane) * 16);
        acc = __builtin_amdgcn_wmma_f32_16x16x32_bf16(false, ua.v, false, bf,
                                                      (short)0, acc, false, false);
      }
      const int col = nt * 16 + l15;
      const float bias = b_gnn[col];                 // layer 0 bias
#pragma unroll
      for (int v = 0; v < 8; ++v) {
        const float p = acc[v];
        const int brow = v + rowh;                   // local batch row
#pragma unroll
        for (int n = 0; n < NVERTS; ++n) {
          float x = s_rowsum[n] * p + bias;
          h_lds[(brow * NVERTS + n) * HIDDEN + col] = (bf16_t)fmaxf(x, 0.f);
        }
      }
    }
  }
  __syncthreads();

  // ---- Phase 2: layers 1..3, h resident in LDS ----------------------------
  for (int l = 1; l < 4; ++l) {
    // Build aggregated A strip (this lane's row, its 128-of-256 K slots).
    v16bf afrag[8];
#pragma unroll
    for (int f = 0; f < 8; ++f) {
      float vals[16];
#pragma unroll
      for (int j = 0; j < 16; ++j) vals[j] = 0.f;
#pragma unroll
      for (int m = 0; m < NVERTS; ++m) {
        const bf16_t* hp = h_lds + (b_own * NVERTS + m) * HIDDEN + f * 32 + khA;
        v8bf c0 = *(const v8bf*)hp;
        v8bf c1 = *(const v8bf*)(hp + 16);
        const float w = adjv[m];
#pragma unroll
        for (int j = 0; j < 8; ++j) {
          vals[j]     += w * (float)c0[j];
          vals[8 + j] += w * (float)c1[j];
        }
      }
      union { v16bf v; bf16_t e[16]; } ua;
#pragma unroll
      for (int j = 0; j < 16; ++j) ua.e[j] = (bf16_t)vals[j];
      afrag[f] = ua.v;
    }
    __syncthreads();   // all waves done READING old h before anyone writes

    const bf16_t* wl = wg_sw + (size_t)l * 65536;
    if (l < 3) __builtin_prefetch(wg_sw + (size_t)(l + 1) * 65536, 0, 1);

#pragma unroll 1
    for (int nt = 0; nt < 16; ++nt) {
      v8f acc = {};
#pragma unroll
      for (int f = 0; f < 8; ++f) {
        const v16bf bf = *(const v16bf*)(wl + ((size_t)(f * 16 + nt) * 32 + lane) * 16);
        acc = __builtin_amdgcn_wmma_f32_16x16x32_bf16(false, afrag[f], false, bf,
                                                      (short)0, acc, false, false);
      }
      const int col = nt * 16 + l15;
      const float bias = b_gnn[l * HIDDEN + col];
#pragma unroll
      for (int v = 0; v < 8; ++v) {
        const int r2 = wave * 16 + v + rowh;
        float x = acc[v] + bias;
        h_lds[r2 * HIDDEN + col] = (bf16_t)fmaxf(x, 0.f);
      }
    }
    __syncthreads();
  }

  // ---- Phase 3: out = template + h @ W_out + b_out (192 rows x 3) ---------
  for (int o = tid; o < MT * NVERTS * 3; o += BLOCK_THR) {
    const int r = o / 3, j = o - r * 3;
    const int bl = r / NVERTS, n = r % NVERTS;
    float s = b_out[j];
    const bf16_t* hp = h_lds + r * HIDDEN;
#pragma unroll 8
    for (int k = 0; k < HIDDEN; ++k) s += (float)hp[k] * W_out[k * 3 + j];
    out[((size_t)(bbase + bl) * NVERTS + n) * 3 + j] = templ[n * 3 + j] + s;
  }
}

// ---------------------------------------------------------------------------
extern "C" void kernel_launch(void* const* d_in, const int* in_sizes, int n_in,
                              void* d_out, int out_size, void* d_ws, size_t ws_size,
                              hipStream_t stream) {
  const float* text   = (const float*)d_in[0];
  const float* W_text = (const float*)d_in[1];
  const float* b_text = (const float*)d_in[2];
  const float* W_gnn  = (const float*)d_in[3];
  const float* b_gnn  = (const float*)d_in[4];
  const float* W_out  = (const float*)d_in[5];
  const float* b_out  = (const float*)d_in[6];
  const float* adj    = (const float*)d_in[7];
  const float* templ  = (const float*)d_in[8];
  float* out = (float*)d_out;

  bf16_t* textbf = (bf16_t*)((char*)d_ws + WS_TEXTBF_OFF);
  bf16_t* wt_sw  = (bf16_t*)((char*)d_ws + WS_WTEXT_OFF);
  bf16_t* wg_sw  = (bf16_t*)((char*)d_ws + WS_WGNN_OFF);

  const int n4 = BATCH * TEXT_DIM / 4;
  cvt_text_kernel<<<(n4 + 255) / 256, 256, 0, stream>>>(text, textbf, n4);

  const int nsw = TEXT_DIM * HIDDEN + 4 * HIDDEN * HIDDEN;
  swizzle_w_kernel<<<(nsw + 255) / 256, 256, 0, stream>>>(W_text, W_gnn, wt_sw, wg_sw);

  mesh_gnn_kernel<<<BATCH / MT, BLOCK_THR, 0, stream>>>(
      textbf, wt_sw, wg_sw, b_text, b_gnn, W_out, b_out, adj, templ, out);
}
